// Locally_Connected_Module_28269474742339
// MI455X (gfx1250) — compile-verified, matrix operand order checked by reading, not run on hardware
//
#include <hip/hip_runtime.h>
#include <math.h>
#include <stdint.h>

// ---------------------------------------------------------------------------
// CDNA5 (gfx1250) implementation of the LC-CNN reference.
// All GEMM-like work on v_wmma_f32_16x16x32_f16 (wave32 WMMA).
// Key trick: K-dimension reordered to k = r*CIN + c (GEMM is invariant under a
// K-permutation applied to both operands) + NHWC f16 activations between
// layers -> A fragments become contiguous 16B b128 loads instead of scalar
// im2col gathers. Conv B tiles stream through LDS with
// global_load_async_to_lds_b128 (ASYNCcnt); fragments via ds_load_b128 pairs.
// ---------------------------------------------------------------------------

typedef __attribute__((ext_vector_type(16))) _Float16 v16h;
typedef __attribute__((ext_vector_type(8)))  _Float16 half8;
typedef __attribute__((ext_vector_type(8)))  float    v8f;

#define BN_EPS 1e-5f
#define NEG_BIG (-3.402823466e38f)
#define USE_ASYNC_LDS 1

__device__ __forceinline__ v8f wmma_f16(v16h a, v16h b, v8f c) {
  // (neg_a, A, neg_b, B, c_mod, C, reuse_a, reuse_b)
  return __builtin_amdgcn_wmma_f32_16x16x32_f16(false, a, false, b, (short)0, c,
                                                false, false);
}

__device__ __forceinline__ v16h cat8(half8 lo, half8 hi) {
  return __builtin_shufflevector(lo, hi, 0, 1, 2, 3, 4, 5, 6, 7, 8, 9, 10, 11,
                                 12, 13, 14, 15);
}

// ---------------------------------------------------------------------------
// LC weight transpose: (1, 32, CIN, OH, OW, 9) f32 -> (P, 32, KP) f16 with
// k = r*CIN + c, KP = K rounded up to 32 (zero padded). Coalesced f16 writes;
// each source weight read exactly once. Gives the LC GEMM 16B-aligned,
// contiguous per-(pos,o) K-rows -> b128 fragment loads.
// ---------------------------------------------------------------------------
__global__ __launch_bounds__(256) void lc_weight_transpose_kernel(
    const float* __restrict__ w, _Float16* __restrict__ wT, int CIN, int OH,
    int OW, int K, int KP) {
  const int idx = (int)(blockIdx.x * blockDim.x + threadIdx.x);
  const int total = OH * OW * 32 * KP;
  if (idx >= total) return;
  const int k = idx % KP;
  const int o = (idx / KP) % 32;
  const int p = idx / (KP * 32);
  float v = 0.f;
  if (k < K) {
    const int r = k / CIN, c = k % CIN;
    const int oy = p / OW, ox = p % OW;
    v = w[(((o * CIN + c) * OH + oy) * OW + ox) * 9 + r];
  }
  wT[idx] = (_Float16)v;
}

// ---------------------------------------------------------------------------
// Conv weight permute: (COUT, CIN, 3, 3) f32 -> (COUT, 9, CIN) f16
// (k = r*CIN + c ordering, matching the NHWC A-side).
// ---------------------------------------------------------------------------
__global__ __launch_bounds__(256) void conv_weight_permute_kernel(
    const float* __restrict__ w, _Float16* __restrict__ wh, int COUT, int CIN) {
  const int K = CIN * 9;
  const int idx = (int)(blockIdx.x * blockDim.x + threadIdx.x);
  if (idx >= COUT * K) return;
  const int n = idx / K;
  const int kr = idx % K;
  const int r = kr / CIN, c = kr % CIN;
  wh[idx] = (_Float16)w[(n * CIN + c) * 9 + r];
}

// ---------------------------------------------------------------------------
// LC layer 1: CIN=3 (K=27 -> KP=32, single K-step), x is NCHW f32 from the
// harness so the A side is a small scalar gather. One wave per position,
// 2x2 grid of 16x16 WMMA tiles (M=batch=32, N=out ch=32).
// Output h1 in NHWC f16.  Fragment layout (16-bit): lane<16 -> K runs
// {0..7,16..23}, lane>=16 -> {8..15,24..31}; M (resp. N) = lane&15.
// ---------------------------------------------------------------------------
__global__ __launch_bounds__(256) void lc1_kernel(
    const float* __restrict__ x, const _Float16* __restrict__ wT,
    const float* __restrict__ bias, const float* __restrict__ act_a,
    _Float16* __restrict__ hout) {
  const int OH = 30, OW = 30, P = 900;
  const int gwave = (int)((blockIdx.x * blockDim.x + threadIdx.x) >> 5);
  if (gwave >= P) return;  // wave-uniform exit
  const int lane = (int)(threadIdx.x & 31);
  const int oy = gwave / OW, ox = gwave % OW;
  const float alpha = act_a[0];
  const int l16 = lane & 15;
  const int koff = (lane & 16) ? 8 : 0;

  const _Float16* brow0 = wT + ((size_t)gwave * 32 + l16) * 32 + koff;
  const _Float16* brow1 = brow0 + (size_t)16 * 32;
  const v16h b0 = cat8(*(const half8*)brow0, *(const half8*)(brow0 + 16));
  const v16h b1 = cat8(*(const half8*)brow1, *(const half8*)(brow1 + 16));

  v16h a[2];
#pragma unroll
  for (int t = 0; t < 2; ++t) {
    const int brw = t * 16 + l16;  // batch row
#pragma unroll
    for (int e = 0; e < 16; ++e) {
      const int k = koff + (e < 8 ? e : e + 8);
      float av = 0.f;
      if (k < 27) {
        const int r = k / 3, c = k % 3;
        av = x[((brw * 3 + c) * 32 + oy + r / 3) * 32 + ox + (r % 3)];
      }
      a[t][e] = (_Float16)av;
    }
  }
  v8f acc[2][2] = {};
  acc[0][0] = wmma_f16(a[0], b0, acc[0][0]);
  acc[0][1] = wmma_f16(a[0], b1, acc[0][1]);
  acc[1][0] = wmma_f16(a[1], b0, acc[1][0]);
  acc[1][1] = wmma_f16(a[1], b1, acc[1][1]);

  const int mhalf = (lane & 16) ? 8 : 0;
  const float bv[2] = {bias[(l16 * OH + oy) * OW + ox],
                       bias[((16 + l16) * OH + oy) * OW + ox]};
#pragma unroll
  for (int mt = 0; mt < 2; ++mt)
#pragma unroll
    for (int nt = 0; nt < 2; ++nt)
#pragma unroll
      for (int r = 0; r < 8; ++r) {
        const int m = mt * 16 + mhalf + r;
        const int n = nt * 16 + l16;
        float v = acc[mt][nt][r] + bv[nt];
        v = (v >= 0.f) ? v : alpha * v;
        hout[(((size_t)m * OH + oy) * OW + ox) * 32 + n] = (_Float16)v;
      }
}

// ---------------------------------------------------------------------------
// LC layers 2/3: CIN=32 (K=288, 9 K-steps = the 9 taps of the 3x3 window).
// NHWC f16 input -> per tap the A fragment is two contiguous b128 loads
// (channels koff..koff+7 and 16+koff..) at (b, oy+r/3, ox+r%3). B rows come
// from the pre-transposed (P, 32, 288) f16 weights. 4 WMMAs per tap.
// Output NHWC f16. No bounds checks (valid convolution).
// ---------------------------------------------------------------------------
__global__ __launch_bounds__(256) void lc_nhwc_kernel(
    const _Float16* __restrict__ hin, const _Float16* __restrict__ wT,
    const float* __restrict__ bias, const float* __restrict__ act_a,
    _Float16* __restrict__ hout, int H) {
  const int W = H, OH = H - 2, OW = W - 2, P = OH * OW;
  const int gwave = (int)((blockIdx.x * blockDim.x + threadIdx.x) >> 5);
  if (gwave >= P) return;  // wave-uniform exit
  const int lane = (int)(threadIdx.x & 31);
  const int oy = gwave / OW, ox = gwave % OW;
  const float alpha = act_a[0];
  const int l16 = lane & 15;
  const int koff = (lane & 16) ? 8 : 0;

  const _Float16* brow0 = wT + ((size_t)gwave * 32 + l16) * 288 + koff;
  const _Float16* brow1 = brow0 + (size_t)16 * 288;
  const _Float16* arow0 = hin + (size_t)l16 * H * W * 32;         // batch l16
  const _Float16* arow1 = hin + (size_t)(16 + l16) * H * W * 32;  // batch 16+l16

  v8f acc[2][2] = {};
#pragma unroll
  for (int r = 0; r < 9; ++r) {
    const size_t po = ((size_t)(oy + r / 3) * W + (ox + r % 3)) * 32 + koff;
    const v16h a0 = cat8(*(const half8*)(arow0 + po),
                         *(const half8*)(arow0 + po + 16));
    const v16h a1 = cat8(*(const half8*)(arow1 + po),
                         *(const half8*)(arow1 + po + 16));
    const int kb = r * 32;
    const v16h b0 = cat8(*(const half8*)(brow0 + kb),
                         *(const half8*)(brow0 + kb + 16));
    const v16h b1 = cat8(*(const half8*)(brow1 + kb),
                         *(const half8*)(brow1 + kb + 16));
    acc[0][0] = wmma_f16(a0, b0, acc[0][0]);
    acc[0][1] = wmma_f16(a0, b1, acc[0][1]);
    acc[1][0] = wmma_f16(a1, b0, acc[1][0]);
    acc[1][1] = wmma_f16(a1, b1, acc[1][1]);
  }

  const int mhalf = (lane & 16) ? 8 : 0;
  const float bv[2] = {bias[(l16 * OH + oy) * OW + ox],
                       bias[((16 + l16) * OH + oy) * OW + ox]};
#pragma unroll
  for (int mt = 0; mt < 2; ++mt)
#pragma unroll
    for (int nt = 0; nt < 2; ++nt)
#pragma unroll
      for (int r = 0; r < 8; ++r) {
        const int m = mt * 16 + mhalf + r;
        const int n = nt * 16 + l16;
        float v = acc[mt][nt][r] + bv[nt];
        v = (v >= 0.f) ? v : alpha * v;
        hout[(((size_t)m * OH + oy) * OW + ox) * 32 + n] = (_Float16)v;
      }
}

// ---------------------------------------------------------------------------
// conv3x3 pad1, implicit GEMM: M = B*H*W (mult of 32), N = COUT (mult of 64),
// K = CIN*9 (mult of 32), k = r*CIN + c. Block (8 waves) owns 32(M) x 64(N).
// Per 32-wide K chunk (single tap r, channel run c0..c0+31):
//  - B tile async-copied to LDS (global_load_async_to_lds_b128, ASYNCcnt)
//  - A tile: 128 threads each do one predicated half8 NHWC load + LDS store
//  - fragments via ds_load_b128 pairs, one WMMA per wave.
// Raw (pre-BN) NCHW f32 output; conv bias omitted (train-mode BN cancels it).
// ---------------------------------------------------------------------------
template <int CIN, int COUT>
__global__ __launch_bounds__(256) void conv3x3_wmma_lds_kernel(
    const _Float16* __restrict__ hin, const _Float16* __restrict__ wh,
    float* __restrict__ out, int B, int H, int W) {
  constexpr int K = CIN * 9;
  constexpr int KSTEPS = K / 32;
  const int HW = H * W;
  const int nBlocksN = COUT / 64;
  const int m0 = (int)(blockIdx.x / nBlocksN) * 32;
  const int n0 = (int)(blockIdx.x % nBlocksN) * 64;

  __shared__ __align__(16) _Float16 Alds[32 * 32];
  __shared__ __align__(16) _Float16 Blds[64 * 32];

  const int tid = (int)threadIdx.x;
  const int lane = tid & 31;
  const int l16 = lane & 15;
  const int koff = (lane & 16) ? 8 : 0;
  const int mt = (tid >> 5) >> 2;  // 0..1
  const int nt = (tid >> 5) & 3;   // 0..3

  // A staging: threads 0..127, one half8 (8 channels of one row) each.
  const int a_row = tid >> 2;        // 0..31 for tid<128
  const int a_kk = (tid & 3) * 8;    // 0,8,16,24
  int ab = 0, ay = 0, ax = 0;
  if (tid < 128) {
    const int am = m0 + a_row;
    ab = am / HW;
    const int arem = am % HW;
    ay = arem / W;
    ax = arem % W;
  }

  // B staging: thread owns 8 consecutive halves of one row (one b128).
  const int b_row = (tid * 8) >> 5;  // 0..63
  const int b_kk = (tid * 8) & 31;   // 0,8,16,24
  const _Float16* bsrc = wh + (size_t)(n0 + b_row) * K + b_kk;
  _Float16* bdst = &Blds[b_row * 32 + b_kk];
#if USE_ASYNC_LDS
  const unsigned bdst_lds = (unsigned)(uintptr_t)bdst;  // wave-relative LDS addr
#endif

  v8f acc = {};
  for (int ks = 0; ks < KSTEPS; ++ks) {
    const int kbase = ks * 32;
    // ---- stage B via async copy to LDS
#if USE_ASYNC_LDS
    asm volatile("global_load_async_to_lds_b128 %0, %1, off"
                 :: "v"(bdst_lds), "v"(bsrc + kbase)
                 : "memory");
#else
    *(half8*)bdst = *(const half8*)(bsrc + kbase);
#endif
    // ---- stage A: one tap (r), channel run c0 + a_kk; validity is uniform
    //      across the whole 8-channel run.
    if (tid < 128) {
      const int r = kbase / CIN;        // CIN,kbase multiples of 32
      const int c0 = kbase % CIN;
      const int iy = ay + r / 3 - 1, ix = ax + (r % 3) - 1;
      half8 av = {};
      if (iy >= 0 && iy < H && ix >= 0 && ix < W)
        av = *(const half8*)(hin + (((size_t)ab * H + iy) * W + ix) * CIN + c0 +
                             a_kk);
      *(half8*)&Alds[a_row * 32 + a_kk] = av;
    }
#if USE_ASYNC_LDS
    asm volatile("s_wait_asynccnt 0x0" ::: "memory");
#endif
    __syncthreads();

    // ---- fragments: two ds_load_b128 each for A and B
    const half8 a_lo = *(const half8*)&Alds[(mt * 16 + l16) * 32 + koff];
    const half8 a_hi = *(const half8*)&Alds[(mt * 16 + l16) * 32 + 16 + koff];
    const half8 b_lo = *(const half8*)&Blds[(nt * 16 + l16) * 32 + koff];
    const half8 b_hi = *(const half8*)&Blds[(nt * 16 + l16) * 32 + 16 + koff];
    acc = wmma_f16(cat8(a_lo, a_hi), cat8(b_lo, b_hi), acc);
    __syncthreads();  // protect LDS before next chunk's staging
  }

  // epilogue: D layout VGPR r -> M = r (+8 for lanes>=16), N = lane&15.
  const int mhalf = (lane & 16) ? 8 : 0;
  const int n = n0 + nt * 16 + l16;
#pragma unroll
  for (int r = 0; r < 8; ++r) {
    const int mm = m0 + mt * 16 + mhalf + r;
    const int bb = mm / HW;
    const int rem = mm % HW;
    out[((bb * COUT + n) * H + (rem / W)) * W + (rem % W)] = acc[r];
  }
}

// ---------------------------------------------------------------------------
// Per-channel batch-norm stats over (B, H, W) of the raw f32 conv output:
// mean + rsqrt(biased var + eps). One block per channel.
// ---------------------------------------------------------------------------
__global__ __launch_bounds__(256) void bn_stats_kernel(
    const float* __restrict__ x, float* __restrict__ mean,
    float* __restrict__ rstd, int B, int C, int HW) {
  const int c = (int)blockIdx.x;
  __shared__ float ss[256], sq[256];
  float s = 0.f, q = 0.f;
  const int N = B * HW;
  for (int i = (int)threadIdx.x; i < N; i += 256) {
    const int b = i / HW, hw = i - b * HW;
    const float v = x[(b * C + c) * HW + hw];
    s += v;
    q += v * v;
  }
  ss[threadIdx.x] = s;
  sq[threadIdx.x] = q;
  __syncthreads();
  for (int st = 128; st > 0; st >>= 1) {
    if ((int)threadIdx.x < st) {
      ss[threadIdx.x] += ss[threadIdx.x + st];
      sq[threadIdx.x] += sq[threadIdx.x + st];
    }
    __syncthreads();
  }
  if (threadIdx.x == 0) {
    const float inv = 1.f / (float)N;
    const float mv = ss[0] * inv;
    const float var = sq[0] * inv - mv * mv;
    mean[c] = mv;
    rstd[c] = rsqrtf(var + BN_EPS);
  }
}

// ---------------------------------------------------------------------------
// Fused BN + PReLU + 2x2 maxpool (VALID): raw NCHW f32 -> NHWC f16.
// c fastest in the output index -> coalesced f16 writes.
// ---------------------------------------------------------------------------
__global__ __launch_bounds__(256) void bn_prelu_pool_kernel(
    const float* __restrict__ x, const float* __restrict__ mean,
    const float* __restrict__ rstd, const float* __restrict__ g,
    const float* __restrict__ beta, const float* __restrict__ act_a,
    _Float16* __restrict__ out, int B, int C, int H, int W, int PH, int PW) {
  const int idx = (int)(blockIdx.x * blockDim.x + threadIdx.x);
  const int total = B * PH * PW * C;
  if (idx >= total) return;
  const int c = idx % C;
  int t = idx / C;
  const int pw = t % PW;
  t /= PW;
  const int ph = t % PH;
  const int b = t / PH;
  const float gm = g[c] * rstd[c];
  const float bt = beta[c] - mean[c] * gm;
  const float a = act_a[0];
  const float* base = x + ((size_t)(b * C + c) * H + ph * 2) * W + pw * 2;
  float best = NEG_BIG;
#pragma unroll
  for (int dy = 0; dy < 2; ++dy)
#pragma unroll
    for (int dx = 0; dx < 2; ++dx) {
      float v = base[dy * W + dx] * gm + bt;
      v = (v >= 0.f) ? v : a * v;
      best = fmaxf(best, v);
    }
  out[idx] = (_Float16)best;
}

// ---------------------------------------------------------------------------
// Final BN + tanh, elementwise (raw NCHW f32 -> NCHW f32 output).
// ---------------------------------------------------------------------------
__global__ __launch_bounds__(256) void bn_tanh_kernel(
    const float* __restrict__ x, const float* __restrict__ mean,
    const float* __restrict__ rstd, const float* __restrict__ g,
    const float* __restrict__ beta, float* __restrict__ out, int B, int C,
    int HW) {
  const int idx = (int)(blockIdx.x * blockDim.x + threadIdx.x);
  if (idx >= B * C * HW) return;
  const int c = (idx / HW) % C;
  const float gm = g[c] * rstd[c];
  const float bt = beta[c] - mean[c] * gm;
  out[idx] = tanhf(x[idx] * gm + bt);
}

// ---------------------------------------------------------------------------
extern "C" void kernel_launch(void* const* d_in, const int* in_sizes, int n_in,
                              void* d_out, int out_size, void* d_ws,
                              size_t ws_size, hipStream_t stream) {
  const float* x    = (const float*)d_in[0];
  const float* lc1w = (const float*)d_in[1];
  const float* lc1b = (const float*)d_in[2];
  const float* lc2w = (const float*)d_in[3];
  const float* lc2b = (const float*)d_in[4];
  const float* lc3w = (const float*)d_in[5];
  const float* lc3b = (const float*)d_in[6];
  const float* acta = (const float*)d_in[7];
  const float* c1w  = (const float*)d_in[8];   // c*_b unused: BN cancels conv bias
  const float* c1g  = (const float*)d_in[10];
  const float* c1be = (const float*)d_in[11];
  const float* c1a  = (const float*)d_in[12];
  const float* c2w  = (const float*)d_in[13];
  const float* c2g  = (const float*)d_in[15];
  const float* c2be = (const float*)d_in[16];
  const float* c2a  = (const float*)d_in[17];
  const float* c3w  = (const float*)d_in[18];
  const float* c3g  = (const float*)d_in[20];
  const float* c3be = (const float*)d_in[21];
  const float* c3a  = (const float*)d_in[22];
  const float* cfw  = (const float*)d_in[23];
  const float* cfg  = (const float*)d_in[25];
  const float* cfbe = (const float*)d_in[26];
  float* outp = (float*)d_out;

  const int B = 32;
  float* ws = (float*)d_ws;
  size_t o = 0;
  float* r1 = ws + o; o += (size_t)B * 64 * 26 * 26;   // conv1 raw (NCHW f32)
  float* r2 = ws + o; o += (size_t)B * 128 * 13 * 13;  // conv2 raw
  float* r3 = ws + o; o += (size_t)B * 256 * 6 * 6;    // conv3 raw
  float* rf = ws + o; o += (size_t)B * 256 * 3 * 3;    // final conv raw
  float* bnm = ws + o; o += 256;                       // mean scratch
  float* bnr = ws + o; o += 256;                       // rstd scratch
  o = (o + 3) & ~(size_t)3;                            // 16B align for f16 region
  _Float16* hreg = (_Float16*)(ws + o);
  size_t ho = 0;
  _Float16* wh1 = hreg + ho; ho += (size_t)64 * 9 * 32;    // conv weights (N,9,CIN) f16
  _Float16* wh2 = hreg + ho; ho += (size_t)128 * 9 * 64;
  _Float16* wh3 = hreg + ho; ho += (size_t)256 * 9 * 128;
  _Float16* whf = hreg + ho; ho += (size_t)256 * 9 * 256;
  _Float16* lt1 = hreg + ho; ho += (size_t)900 * 32 * 32;   // lc wT (KP-padded)
  _Float16* lt2 = hreg + ho; ho += (size_t)784 * 32 * 288;
  _Float16* lt3 = hreg + ho; ho += (size_t)676 * 32 * 288;
  _Float16* h1  = hreg + ho; ho += (size_t)B * 30 * 30 * 32;  // NHWC f16 acts
  _Float16* h2  = hreg + ho; ho += (size_t)B * 28 * 28 * 32;
  _Float16* h3  = hreg + ho; ho += (size_t)B * 26 * 26 * 32;
  _Float16* p1  = hreg + ho; ho += (size_t)B * 13 * 13 * 64;
  _Float16* p2  = hreg + ho; ho += (size_t)B * 6 * 6 * 128;
  _Float16* p3  = hreg + ho; ho += (size_t)B * 3 * 3 * 256;

  // ---- weight preprocessing: conv permute->f16, LC transpose->f16
  conv_weight_permute_kernel<<<(64 * 288 + 255) / 256, 256, 0, stream>>>(c1w, wh1, 64, 32);
  conv_weight_permute_kernel<<<(128 * 576 + 255) / 256, 256, 0, stream>>>(c2w, wh2, 128, 64);
  conv_weight_permute_kernel<<<(256 * 1152 + 255) / 256, 256, 0, stream>>>(c3w, wh3, 256, 128);
  conv_weight_permute_kernel<<<(256 * 2304 + 255) / 256, 256, 0, stream>>>(cfw, whf, 256, 256);
  {
    int tot = 900 * 32 * 32;
    lc_weight_transpose_kernel<<<(tot + 255) / 256, 256, 0, stream>>>(
        lc1w, lt1, 3, 30, 30, 27, 32);
    tot = 784 * 32 * 288;
    lc_weight_transpose_kernel<<<(tot + 255) / 256, 256, 0, stream>>>(
        lc2w, lt2, 32, 28, 28, 288, 288);
    tot = 676 * 32 * 288;
    lc_weight_transpose_kernel<<<(tot + 255) / 256, 256, 0, stream>>>(
        lc3w, lt3, 32, 26, 26, 288, 288);
  }

  // ---- locally-connected stack (1 wave per spatial position, 8 waves/block)
  lc1_kernel<<<(900 + 7) / 8, 256, 0, stream>>>(x, lt1, lc1b, acta, h1);
  lc_nhwc_kernel<<<(784 + 7) / 8, 256, 0, stream>>>(h1, lt2, lc2b, acta, h2, 30);
  lc_nhwc_kernel<<<(676 + 7) / 8, 256, 0, stream>>>(h2, lt3, lc3b, acta, h3, 28);

  // ---- block 1: conv 32->64 @26, BN, PReLU, pool -> 13
  {
    const int M = B * 26 * 26;
    conv3x3_wmma_lds_kernel<32, 64><<<(M / 32) * (64 / 64), 256, 0, stream>>>(h3, wh1, r1, B, 26, 26);
    bn_stats_kernel<<<64, 256, 0, stream>>>(r1, bnm, bnr, B, 64, 26 * 26);
    const int tot = B * 64 * 13 * 13;
    bn_prelu_pool_kernel<<<(tot + 255) / 256, 256, 0, stream>>>(
        r1, bnm, bnr, c1g, c1be, c1a, p1, B, 64, 26, 26, 13, 13);
  }
  // ---- block 2: conv 64->128 @13, BN, PReLU, pool -> 6
  {
    const int M = B * 13 * 13;
    conv3x3_wmma_lds_kernel<64, 128><<<(M / 32) * (128 / 64), 256, 0, stream>>>(p1, wh2, r2, B, 13, 13);
    bn_stats_kernel<<<128, 256, 0, stream>>>(r2, bnm, bnr, B, 128, 13 * 13);
    const int tot = B * 128 * 6 * 6;
    bn_prelu_pool_kernel<<<(tot + 255) / 256, 256, 0, stream>>>(
        r2, bnm, bnr, c2g, c2be, c2a, p2, B, 128, 13, 13, 6, 6);
  }
  // ---- block 3: conv 128->256 @6, BN, PReLU, pool -> 3
  {
    const int M = B * 6 * 6;
    conv3x3_wmma_lds_kernel<128, 256><<<(M / 32) * (256 / 64), 256, 0, stream>>>(p2, wh3, r3, B, 6, 6);
    bn_stats_kernel<<<256, 256, 0, stream>>>(r3, bnm, bnr, B, 256, 6 * 6);
    const int tot = B * 256 * 3 * 3;
    bn_prelu_pool_kernel<<<(tot + 255) / 256, 256, 0, stream>>>(
        r3, bnm, bnr, c3g, c3be, c3a, p3, B, 256, 6, 6, 3, 3);
  }
  // ---- final: conv 256->256 @3, BN, tanh
  {
    const int M = B * 3 * 3;
    conv3x3_wmma_lds_kernel<256, 256><<<(M / 32) * (256 / 64), 256, 0, stream>>>(p3, whf, rf, B, 3, 3);
    bn_stats_kernel<<<256, 256, 0, stream>>>(rf, bnm, bnr, B, 256, 3 * 3);
    const int tot = B * 256 * 3 * 3;
    bn_tanh_kernel<<<(tot + 255) / 256, 256, 0, stream>>>(rf, bnm, bnr, cfg, cfbe, outp, B, 256, 3 * 3);
  }
  (void)in_sizes; (void)n_in; (void)out_size; (void)ws_size;
}
